// GCLSTM_RGCN_89008902243176
// MI455X (gfx1250) — matrix operand order repaired and checked by reading, not compiled, not run on hardware
//
#include <hip/hip_runtime.h>
#include <stdint.h>

// ---------------------------------------------------------------------------
// GCLSTM + GCN link predictor forward, CDNA5 (gfx1250), wave32 WMMA bf16.
// N=50000 (divisible by 16), F=64, H=128, CLS=64, gates stacked (i,f,c,o).
// Weights are pre-swizzled into WMMA B-fragment order (one v16bf per lane per
// k-tile). A-tiles for the fused gate GEMM are staged into LDS with
// global_load_async_to_lds_b128 (ASYNCcnt) instead of load+ds_store.
// ---------------------------------------------------------------------------

typedef __attribute__((ext_vector_type(16))) __bf16 v16bf;
typedef __attribute__((ext_vector_type(8)))  float  v8f;

#define F_DIM 64
#define H_DIM 128
#define CLS_DIM 64
#define G_DIM 512   // 4*H

__device__ __forceinline__ float sigm(float x) { return 1.0f / (1.0f + __expf(-x)); }

__device__ __forceinline__ float4 relu4(float4 p) {
    p.x = fmaxf(p.x, 0.f); p.y = fmaxf(p.y, 0.f);
    p.z = fmaxf(p.z, 0.f); p.w = fmaxf(p.w, 0.f);
    return p;
}

// Pack 16 floats (two contiguous 8-float runs) into a WMMA A fragment.
__device__ __forceinline__ v16bf pack_a(float4 p0, float4 p1, float4 p2, float4 p3) {
    v16bf a;
    a[0]  = (__bf16)p0.x; a[1]  = (__bf16)p0.y; a[2]  = (__bf16)p0.z; a[3]  = (__bf16)p0.w;
    a[4]  = (__bf16)p1.x; a[5]  = (__bf16)p1.y; a[6]  = (__bf16)p1.z; a[7]  = (__bf16)p1.w;
    a[8]  = (__bf16)p2.x; a[9]  = (__bf16)p2.y; a[10] = (__bf16)p2.z; a[11] = (__bf16)p2.w;
    a[12] = (__bf16)p3.x; a[13] = (__bf16)p3.y; a[14] = (__bf16)p3.z; a[15] = (__bf16)p3.w;
    return a;
}

// --- weight pack: f32 [K,Nc] row-major -> bf16 WMMA B-fragment order --------
// dst[((ct*ktiles + kt)*32 + lane)*16 + i] = src[kt*32 + (i>>3)*16 + (lane>>4)*8 + (i&7)][ct*16 + (lane&15)]
__global__ void k_pack_b(const float* __restrict__ src, __bf16* __restrict__ dst, int K, int Nc) {
    int idx = blockIdx.x * blockDim.x + threadIdx.x;
    int ktiles = K >> 5;
    int total = (Nc >> 4) * ktiles * 512;
    if (idx >= total) return;
    int i = idx & 15;
    int l = (idx >> 4) & 31;
    int t = idx >> 9;
    int kt = t % ktiles;
    int ct = t / ktiles;
    int k = kt * 32 + ((i >> 3) << 4) + ((l >> 4) << 3) + (i & 7);
    int n = ct * 16 + (l & 15);
    dst[idx] = (__bf16)src[(size_t)k * Nc + n];
}

// --- in-degree (dst side), then dinv = rsqrt(deg+1) ------------------------
__global__ void k_degree(const int* __restrict__ dst_idx, int* __restrict__ deg, int E) {
    int i = blockIdx.x * blockDim.x + threadIdx.x;
    if (i < E) atomicAdd(&deg[dst_idx[i]], 1);
}
__global__ void k_dinv(const int* __restrict__ deg, float* __restrict__ dinv, int N) {
    int i = blockIdx.x * blockDim.x + threadIdx.x;
    if (i < N) dinv[i] = rsqrtf((float)deg[i] + 1.0f);
}

// --- generic small GEMM: D[M,Nc] = (relu?)A[M,K] * Bpacked + bias ----------
// One wave computes one 16x16 D tile. grid = (M/16, Nc/16), block = 32.
template<int RELU, int HAS_BIAS>
__global__ __launch_bounds__(32) void k_gemm_small(
    const float* __restrict__ A, const v16bf* __restrict__ Bf,
    const float* __restrict__ bias, float* __restrict__ D,
    int K, int Nc)
{
    const int ktiles = K >> 5;
    const int rowBase = blockIdx.x * 16;
    const int ct = blockIdx.y;
    const int lane = threadIdx.x;
    const int lrow = lane & 15;
    const int lhi  = lane >> 4;
    const float4* Arow = (const float4*)(A + (size_t)(rowBase + lrow) * K);

    v8f acc = {};
    for (int kt = 0; kt < ktiles; ++kt) {
        const int f4 = (kt << 3) + (lhi << 1);
        float4 p0 = Arow[f4 + 0], p1 = Arow[f4 + 1];
        float4 p2 = Arow[f4 + 4], p3 = Arow[f4 + 5];
        if (RELU) { p0 = relu4(p0); p1 = relu4(p1); p2 = relu4(p2); p3 = relu4(p3); }
        v16bf a = pack_a(p0, p1, p2, p3);
        v16bf b = Bf[(size_t)(ct * ktiles + kt) * 32 + lane];
        acc = __builtin_amdgcn_wmma_f32_16x16x32_bf16(false, a, false, b, (short)0, acc, false, false);
    }
    const int j = ct * 16 + lrow;
    float bv = 0.0f;
    if (HAS_BIAS) bv = bias[j];
#pragma unroll
    for (int e = 0; e < 8; ++e) {
        int m = rowBase + (lhi << 3) + e;
        D[(size_t)m * Nc + j] = acc[e] + bv;
    }
}

// --- GCN: self-loop + bias init, then edge scatter -------------------------
__global__ void k_agg_init(const float* __restrict__ xw, const float* __restrict__ dinv,
                           const float* __restrict__ b_gcn, float* __restrict__ agg, int N) {
    int t = blockIdx.x * blockDim.x + threadIdx.x;
    if (t < N * F_DIM) {
        int n = t >> 6, f = t & 63;
        float d = dinv[n];
        agg[t] = d * d * xw[t] + b_gcn[f];
    }
}
__global__ void k_scatter(const int* __restrict__ ei, const float* __restrict__ xw,
                          const float* __restrict__ dinv, float* __restrict__ agg, int E) {
    int t = blockIdx.x * blockDim.x + threadIdx.x;
    if (t >= E * 16) return;
    int e = t >> 4;
    int f = (t & 15) * 4;
    int s = ei[e];
    int d = ei[E + e];
    float norm = dinv[s] * dinv[d];
    const float* xs = xw + (size_t)s * F_DIM + f;
    float* ad = agg + (size_t)d * F_DIM + f;
#pragma unroll
    for (int u = 0; u < 4; ++u) atomicAdd(&ad[u], norm * xs[u]);
}

// --- fused gate GEMM + LSTM cell ------------------------------------------
// Block = 16 rows x 512 gate-cols; 8 waves, wave w owns j-tile [16w,16w+16)
// across all 4 gates. A tiles (contiguous 4KB + 8KB) staged to LDS via
// global_load_async_to_lds_b128; g never hits memory.
__global__ __launch_bounds__(256) void k_gates(
    const float* __restrict__ h,        // [N,64]
    const float* __restrict__ hidden1,  // [N,128]
    const float* __restrict__ hidden2,  // [N,128]
    const v16bf* __restrict__ Wxf,      // packed [64,512]
    const v16bf* __restrict__ Thf,      // packed [128,512]
    const float* __restrict__ b_gate,   // [512]
    const float* __restrict__ b_conv,   // [512]
    const float* __restrict__ w_c,      // [3,128]
    float* __restrict__ Cout,           // [N,128]
    float* __restrict__ Hn)             // [N,128]
{
    __shared__ __align__(16) float sF0[16 * 64];    // h tile (f32)
    __shared__ __align__(16) float sF1[16 * 128];   // hidden1 tile (f32)

    const int rowBase = blockIdx.x * 16;
    const int tid = threadIdx.x;

    // Async-copy both tiles (contiguous in global memory) into LDS.
    {
        unsigned long long g0 = (unsigned long long)(uintptr_t)(h + (size_t)rowBase * 64)
                              + (unsigned)tid * 16u;
        unsigned la0 = (unsigned)(uintptr_t)sF0 + (unsigned)tid * 16u;
        asm volatile("global_load_async_to_lds_b128 %0, %1, off"
                     :: "v"(la0), "v"(g0) : "memory");

        unsigned long long g1 = (unsigned long long)(uintptr_t)(hidden1 + (size_t)rowBase * 128);
        unsigned lb = (unsigned)(uintptr_t)sF1;
        unsigned long long ga = g1 + (unsigned)tid * 16u;
        unsigned lba = lb + (unsigned)tid * 16u;
        asm volatile("global_load_async_to_lds_b128 %0, %1, off"
                     :: "v"(lba), "v"(ga) : "memory");
        unsigned long long gc = g1 + (unsigned)(tid + 256) * 16u;
        unsigned lbc = lb + (unsigned)(tid + 256) * 16u;
        asm volatile("global_load_async_to_lds_b128 %0, %1, off"
                     :: "v"(lbc), "v"(gc) : "memory");

        asm volatile("s_wait_asynccnt 0x0" ::: "memory");
    }
    __syncthreads();

    const int lane = tid & 31;
    const int lrow = lane & 15;
    const int lhi  = lane >> 4;
    const int ctj  = tid >> 5;          // j-tile index within [0,8)
    const int colBase = ctj * 16;

    const float4* Arow0 = (const float4*)(sF0 + lrow * 64);
    const float4* Arow1 = (const float4*)(sF1 + lrow * 128);

    v8f acc0 = {}, acc1 = {}, acc2 = {}, acc3 = {};

    // K-loop part 1: h @ Wx  (K=64 -> 2 k-tiles, ktiles=2 in pack order)
#pragma unroll
    for (int kt = 0; kt < 2; ++kt) {
        const int f4 = (kt << 3) + (lhi << 1);
        v16bf a = pack_a(Arow0[f4], Arow0[f4 + 1], Arow0[f4 + 4], Arow0[f4 + 5]);
#pragma unroll
        for (int g = 0; g < 4; ++g) {
            v16bf b = Wxf[(size_t)(((g * 8 + ctj) * 2) + kt) * 32 + lane];
            if (g == 0) acc0 = __builtin_amdgcn_wmma_f32_16x16x32_bf16(false, a, false, b, (short)0, acc0, false, false);
            if (g == 1) acc1 = __builtin_amdgcn_wmma_f32_16x16x32_bf16(false, a, false, b, (short)0, acc1, false, false);
            if (g == 2) acc2 = __builtin_amdgcn_wmma_f32_16x16x32_bf16(false, a, false, b, (short)0, acc2, false, false);
            if (g == 3) acc3 = __builtin_amdgcn_wmma_f32_16x16x32_bf16(false, a, false, b, (short)0, acc3, false, false);
        }
    }
    // K-loop part 2: hidden1 @ Th  (K=128 -> 4 k-tiles)
#pragma unroll
    for (int kt = 0; kt < 4; ++kt) {
        const int f4 = (kt << 3) + (lhi << 1);
        v16bf a = pack_a(Arow1[f4], Arow1[f4 + 1], Arow1[f4 + 4], Arow1[f4 + 5]);
#pragma unroll
        for (int g = 0; g < 4; ++g) {
            v16bf b = Thf[(size_t)(((g * 8 + ctj) * 4) + kt) * 32 + lane];
            if (g == 0) acc0 = __builtin_amdgcn_wmma_f32_16x16x32_bf16(false, a, false, b, (short)0, acc0, false, false);
            if (g == 1) acc1 = __builtin_amdgcn_wmma_f32_16x16x32_bf16(false, a, false, b, (short)0, acc1, false, false);
            if (g == 2) acc2 = __builtin_amdgcn_wmma_f32_16x16x32_bf16(false, a, false, b, (short)0, acc2, false, false);
            if (g == 3) acc3 = __builtin_amdgcn_wmma_f32_16x16x32_bf16(false, a, false, b, (short)0, acc3, false, false);
        }
    }

    // LSTM epilogue in registers.  D layout: M = e + 8*lhi, N = lrow.
    const int j = colBase + lrow;
    const float bg0 = b_gate[0 * 128 + j] + b_conv[0 * 128 + j];
    const float bg1 = b_gate[1 * 128 + j] + b_conv[1 * 128 + j];
    const float bg2 = b_gate[2 * 128 + j] + b_conv[2 * 128 + j];
    const float bg3 = b_gate[3 * 128 + j] + b_conv[3 * 128 + j];
    const float wc0 = w_c[0 * 128 + j];
    const float wc1 = w_c[1 * 128 + j];
    const float wc2 = w_c[2 * 128 + j];
#pragma unroll
    for (int e = 0; e < 8; ++e) {
        const int m = rowBase + (lhi << 3) + e;
        const float h2 = hidden2[(size_t)m * H_DIM + j];
        const float gi = acc0[e] + bg0;
        const float gf = acc1[e] + bg1;
        const float gc = acc2[e] + bg2;
        const float go = acc3[e] + bg3;
        const float I  = sigm(gi + wc0 * h2);
        const float Fg = sigm(gf + wc1 * h2);
        const float Cv = Fg * h2 + I * tanhf(gc);
        const float O  = sigm(go + wc2 * Cv);
        Cout[(size_t)m * H_DIM + j] = Cv;
        Hn[(size_t)m * H_DIM + j]   = O * tanhf(Cv);
    }
}

// --- row softmax over 64 classes, in place --------------------------------
__global__ void k_softmax(float* __restrict__ z, int N) {
    int n = blockIdx.x * blockDim.x + threadIdx.x;
    if (n >= N) return;
    float* row = z + (size_t)n * CLS_DIM;
    float mx = row[0];
#pragma unroll 8
    for (int j = 1; j < CLS_DIM; ++j) mx = fmaxf(mx, row[j]);
    float s = 0.0f;
    float v[CLS_DIM];
#pragma unroll 8
    for (int j = 0; j < CLS_DIM; ++j) { v[j] = __expf(row[j] - mx); s += v[j]; }
    float inv = 1.0f / s;
#pragma unroll 8
    for (int j = 0; j < CLS_DIM; ++j) row[j] = v[j] * inv;
}

// --- decode: r[e] = dot(z[src], z[dst]) over 64 ----------------------------
__global__ void k_decode(const int* __restrict__ eli, const float* __restrict__ z,
                         float* __restrict__ r, int EL) {
    int e = blockIdx.x * blockDim.x + threadIdx.x;
    if (e >= EL) return;
    int s = eli[e];
    int d = eli[EL + e];
    const float4* zs = reinterpret_cast<const float4*>(z + (size_t)s * CLS_DIM);
    const float4* zd = reinterpret_cast<const float4*>(z + (size_t)d * CLS_DIM);
    float acc = 0.0f;
#pragma unroll
    for (int q = 0; q < 16; ++q) {
        float4 a = zs[q], b = zd[q];
        acc += a.x * b.x + a.y * b.y + a.z * b.z + a.w * b.w;
    }
    r[e] = acc;
}

// ---------------------------------------------------------------------------
extern "C" void kernel_launch(void* const* d_in, const int* in_sizes, int n_in,
                              void* d_out, int out_size, void* d_ws, size_t ws_size,
                              hipStream_t stream) {
    const float* x        = (const float*)d_in[0];
    const int*   ei       = (const int*)  d_in[1];
    const int*   eli      = (const int*)  d_in[2];
    const float* hidden1  = (const float*)d_in[3];
    const float* hidden2  = (const float*)d_in[4];
    const float* W_gcn    = (const float*)d_in[5];
    const float* b_gcn    = (const float*)d_in[6];
    const float* Wx       = (const float*)d_in[7];
    const float* Th       = (const float*)d_in[8];
    const float* b_gate   = (const float*)d_in[9];
    const float* b_conv   = (const float*)d_in[10];
    const float* w_c      = (const float*)d_in[11];
    const float* W_lin    = (const float*)d_in[12];
    const float* b_lin    = (const float*)d_in[13];

    const int N  = in_sizes[0] / F_DIM;   // 50000
    const int E  = in_sizes[1] / 2;       // 800000
    const int EL = in_sizes[2] / 2;       // 200000

    // workspace carve-up (float elements; all bf16 regions 32B aligned)
    float* ws    = (float*)d_ws;
    float* dinv  = ws;                                  // N
    int*   deg   = (int*)(ws + N);                      // N
    float* xw    = ws + 2 * (size_t)N;                  // N*64
    float* agg   = xw + (size_t)N * F_DIM;              // N*64  (== h)
    float* Hn    = agg + (size_t)N * F_DIM;             // N*128
    float* z     = Hn + (size_t)N * H_DIM;              // N*64 (logits -> softmax in place)
    __bf16* Wgcnp = (__bf16*)(z + (size_t)N * CLS_DIM); // 64*64   packed
    __bf16* Wxp   = Wgcnp + F_DIM * F_DIM;              // 64*512  packed
    __bf16* Thp   = Wxp + F_DIM * G_DIM;                // 128*512 packed
    __bf16* Wlinp = Thp + H_DIM * G_DIM;                // 128*64  packed

    // outputs: r | hidden1 | C
    float* out_r  = (float*)d_out;
    float* out_h1 = out_r + EL;
    float* out_C  = out_h1 + (size_t)N * H_DIM;

    const int T = 256;
    // 1) pack weights to bf16 B-fragment order
    k_pack_b<<<(F_DIM * F_DIM + T - 1) / T, T, 0, stream>>>(W_gcn, Wgcnp, F_DIM, F_DIM);
    k_pack_b<<<(F_DIM * G_DIM + T - 1) / T, T, 0, stream>>>(Wx, Wxp, F_DIM, G_DIM);
    k_pack_b<<<(H_DIM * G_DIM + T - 1) / T, T, 0, stream>>>(Th, Thp, H_DIM, G_DIM);
    k_pack_b<<<(H_DIM * CLS_DIM + T - 1) / T, T, 0, stream>>>(W_lin, Wlinp, H_DIM, CLS_DIM);

    // 2) degrees -> dinv
    hipMemsetAsync(deg, 0, (size_t)N * sizeof(int), stream);
    k_degree<<<(E + T - 1) / T, T, 0, stream>>>(ei + E, deg, E);
    k_dinv<<<(N + T - 1) / T, T, 0, stream>>>(deg, dinv, N);

    // 3) xw = x @ W_gcn  (WMMA)
    k_gemm_small<0, 0><<<dim3(N / 16, F_DIM / 16), 32, 0, stream>>>(
        x, (const v16bf*)Wgcnp, nullptr, xw, F_DIM, F_DIM);

    // 4) GCN aggregate: self-loop + bias, then edge scatter
    k_agg_init<<<((size_t)N * F_DIM + T - 1) / T, T, 0, stream>>>(xw, dinv, b_gcn, agg, N);
    k_scatter<<<((size_t)E * 16 + T - 1) / T, T, 0, stream>>>(ei, xw, dinv, agg, E);

    // 5) fused gate GEMMs + LSTM cell (WMMA + async-to-LDS staging)
    k_gates<<<N / 16, 256, 0, stream>>>(agg, hidden1, hidden2,
                                        (const v16bf*)Wxp, (const v16bf*)Thp,
                                        b_gate, b_conv, w_c, out_C, Hn);

    // 6) logits = relu(Hn) @ W_lin + b_lin (WMMA), softmax in place
    k_gemm_small<1, 1><<<dim3(N / 16, CLS_DIM / 16), 32, 0, stream>>>(
        Hn, (const v16bf*)Wlinp, b_lin, z, H_DIM, CLS_DIM);
    k_softmax<<<(N + T - 1) / T, T, 0, stream>>>(z, N);

    // 7) decode edge scores
    k_decode<<<(EL + T - 1) / T, T, 0, stream>>>(eli, z, out_r, EL);

    // 8) hidden1 passthrough output
    hipMemcpyAsync(out_h1, hidden1, (size_t)N * H_DIM * sizeof(float),
                   hipMemcpyDeviceToDevice, stream);
}
